// pLayer_4621384811111
// MI455X (gfx1250) — matrix operand order repaired and checked by reading, not compiled, not run on hardware
//
#include <hip/hip_runtime.h>

#define GMIN 0.01f
#define GMAX 10.0f

typedef __attribute__((ext_vector_type(16))) _Float16 v16h;
typedef __attribute__((ext_vector_type(8)))  float    v8f;

// Problem geometry (fixed by the reference)
#define BTOT   32768
#define NIN    256
#define NOUT   256
#define KEXT   258          // n_in + 2
#define KTOT   516          // 2*(n_in+2)
#define KPAD   544          // padded to 17 * 32
#define NKC    17           // K chunks of 32
#define BM     32           // rows per block (2 M-tiles of 16)

__device__ __forceinline__ float ptanh_dev(float x, float e0, float e1, float e2, float e3) {
    return e0 + e1 * tanhf((x - e2) * e3);
}

// LDS index for A-fragment layout (16-bit A 16x32, wave32):
//   lanes 0-15 : M=lane,    halves 0..7 = K 0..7,  halves 8..15 = K 16..23
//   lanes 16-31: M=lane-16, halves 0..7 = K 8..15, halves 8..15 = K 24..31
// Fragments stored as [kc][mt][lane][16 halves].
__device__ __forceinline__ int aIdx(int m, int k) {
    int kc = k >> 5, kk = k & 31;
    int mt = m >> 4, mr = m & 15;
    int lane = mr + ((kk & 8) ? 16 : 0);
    int pos  = (kk & 7) + ((kk & 16) ? 8 : 0);
    return ((kc * 2 + mt) * 32 + lane) * 16 + pos;
}

// ---------------------------------------------------------------------------
// Kernel 1: fold clip/threshold/normalize/sign-split of theta into a padded
// [KPAD x 256] f16 weight matrix, stored directly in WMMA B-fragment layout:
//   fragment f = kc*16 + nt  (32x16 K x N tile), 32 lanes x 16 halves:
//   lane l<16 : N = nt*16+l,    halves 0..7 = K kc*32+0..7, halves 8..15 = K kc*32+16..23
//   lane l>=16: N = nt*16+l-16, halves 0..7 = K kc*32+8..15, halves 8..15 = K kc*32+24..31
// ---------------------------------------------------------------------------
__global__ __launch_bounds__(256) void pack_weights(const float* __restrict__ theta,
                                                    _Float16* __restrict__ Wb) {
    __shared__ float colsum[NOUT];
    const int tid = threadIdx.x;   // 256 threads, single block

    float s = 0.0f;
    for (int i = 0; i < KEXT; ++i) {
        float t = theta[i * NOUT + tid];
        t = fminf(fmaxf(t, -GMAX), GMAX);
        float at = fabsf(t);
        if (at < GMIN) at = 0.0f;
        s += at;
    }
    colsum[tid] = s;
    __syncthreads();

    for (int idx = tid; idx < KPAD * NOUT; idx += 256) {
        int k = idx >> 8;          // 0..543
        int o = idx & 255;         // 0..255
        float val = 0.0f;
        if (k < KTOT) {
            int i = (k < KEXT) ? k : (k - KEXT);
            float t = theta[i * NOUT + o];
            t = fminf(fmaxf(t, -GMAX), GMAX);
            float at = fabsf(t);
            if (at < GMIN) { t = 0.0f; at = 0.0f; }
            float w = (at > 0.0f) ? (at / colsum[o]) : 0.0f;
            bool pos = (t >= 0.0f);
            val = (k < KEXT) ? (pos ? w : 0.0f) : (pos ? 0.0f : w);
        }
        int kc = k >> 5, kk = k & 31, nt = o >> 4, nl = o & 15;
        int lane = nl + ((kk & 8) ? 16 : 0);
        int pos  = (kk & 7) + ((kk & 16) ? 8 : 0);
        Wb[((kc * 16 + nt) * 32 + lane) * 16 + pos] = (_Float16)val;
    }
}

// ---------------------------------------------------------------------------
// Kernel 2: fused  ptanh-expand -> GEMM (WMMA f16->f32) -> ptanh epilogue.
// Block: 256 threads (8 waves), covers BM=32 rows x all 256 columns.
// Wave grid: mt in {0,1} (M tile), ng in {0..3} (4 N-subtiles each).
// ---------------------------------------------------------------------------
__global__ __launch_bounds__(256) void plant_gemm(const float* __restrict__ a,
                                                  const _Float16* __restrict__ Wb,
                                                  const float* __restrict__ eta_act,
                                                  const float* __restrict__ eta_inv,
                                                  float* __restrict__ out) {
    __shared__ _Float16 aLds[NKC * 2 * 512] __attribute__((aligned(32))); // 34,816 B

    const int tid     = threadIdx.x;
    const int block_m = blockIdx.x * BM;

    const float i0 = eta_inv[0], i1 = eta_inv[1], i2 = eta_inv[2], i3 = eta_inv[3];

    // ---- Stage A: each a element loaded once, tanh'd once, written as two
    //      f16 K-slots (k=j for a_ext, k=KEXT+j for a_neg) in fragment layout.
    for (int m = 0; m < BM; ++m) {
        float v  = a[(size_t)(block_m + m) * NIN + tid];           // coalesced
        float vn = ptanh_dev(v, i0, i1, i2, i3);
        aLds[aIdx(m, tid)]        = (_Float16)v;
        aLds[aIdx(m, KEXT + tid)] = (_Float16)vn;
    }
    // Constant/padding K slots: k=256 ->1, k=257 ->0, k=514 ->ptanh(1), k=515 ->0, k>=516 ->0
    const float onev = ptanh_dev(1.0f, i0, i1, i2, i3);
    for (int idx = tid; idx < BM * 32; idx += 256) {
        int m = idx >> 5, p = idx & 31;
        int k; float val;
        if      (p == 0) { k = 256; val = 1.0f; }
        else if (p == 1) { k = 257; val = 0.0f; }
        else if (p == 2) { k = 514; val = onev; }
        else if (p == 3) { k = 515; val = 0.0f; }
        else             { k = 516 + (p - 4); val = 0.0f; }
        aLds[aIdx(m, k)] = (_Float16)val;
    }
    __syncthreads();

    const int wave = tid >> 5;
    const int lane = tid & 31;
    const int mt   = wave & 1;      // M tile within block
    const int ng   = wave >> 1;     // N group (4 subtiles of 16)

    v8f acc[4];
    #pragma unroll
    for (int i = 0; i < 4; ++i) acc[i] = (v8f)0.0f;

    for (int kc = 0; kc < NKC; ++kc) {
        // A fragment: one contiguous 32-byte per-lane slab from LDS
        v16h afrag = *(const v16h*)&aLds[((kc * 2 + mt) * 32 + lane) * 16];
        const _Float16* bbase = Wb + ((size_t)(kc * 16 + ng * 4) * 32 + lane) * 16;
        #pragma unroll
        for (int ns = 0; ns < 4; ++ns) {
            v16h bfrag = *(const v16h*)(bbase + (size_t)ns * 512);
            acc[ns] = __builtin_amdgcn_wmma_f32_16x16x32_f16(
                false, afrag, false, bfrag, (short)0, acc[ns], false, false);
        }
    }

    // ---- Epilogue: ptanh(eta_act) + coalesced f32 stores.
    // C layout: VGPR r -> M=r (lanes 0-15, N=lane) / M=r+8 (lanes 16-31, N=lane-16)
    const float e0 = eta_act[0], e1 = eta_act[1], e2 = eta_act[2], e3 = eta_act[3];
    const int row_base = block_m + mt * 16 + ((lane < 16) ? 0 : 8);
    const int nl = lane & 15;
    #pragma unroll
    for (int ns = 0; ns < 4; ++ns) {
        int col = (ng * 4 + ns) * 16 + nl;
        #pragma unroll
        for (int r = 0; r < 8; ++r) {
            float z = acc[ns][r];
            out[(size_t)(row_base + r) * NOUT + col] = ptanh_dev(z, e0, e1, e2, e3);
        }
    }
}

// ---------------------------------------------------------------------------
extern "C" void kernel_launch(void* const* d_in, const int* in_sizes, int n_in,
                              void* d_out, int out_size, void* d_ws, size_t ws_size,
                              hipStream_t stream) {
    const float* a        = (const float*)d_in[0];  // [32768, 256]
    const float* theta    = (const float*)d_in[1];  // [258, 256]
    const float* eta_act  = (const float*)d_in[2];  // [4]
    const float* eta_inv  = (const float*)d_in[3];  // [4]
    float*       out      = (float*)d_out;          // [32768, 256]
    _Float16*    Wb       = (_Float16*)d_ws;        // 544*256*2 = 278,528 B

    pack_weights<<<1, 256, 0, stream>>>(theta, Wb);
    plant_gemm<<<BTOT / BM, 256, 0, stream>>>(a, Wb, eta_act, eta_inv, out);
}
